// TransformerBlock_90091234001512
// MI455X (gfx1250) — matrix operand order repaired and checked by reading, not compiled, not run on hardware
//
#include <hip/hip_runtime.h>
#include <hip/hip_bf16.h>
#include <math.h>

typedef __bf16 bf16;
typedef __attribute__((ext_vector_type(16))) __bf16 v16bf;
typedef __attribute__((ext_vector_type(8)))  float  v8f;
typedef __attribute__((ext_vector_type(4)))  unsigned int v4u;

#define WMMA_BF16(A_, B_, C_) \
  __builtin_amdgcn_wmma_f32_16x16x32_bf16(false, (A_), false, (B_), (short)0, (C_), false, false)

static __device__ __forceinline__ v16bf ld_frag_g(const bf16* p0, const bf16* p1) {
  union { v4u u[2]; v16bf v; } r;
  r.u[0] = *(const v4u*)p0;
  r.u[1] = *(const v4u*)p1;
  return r.v;
}
static __device__ __forceinline__ v16bf zfrag() {
  union { unsigned int u[8]; v16bf v; } r;
  #pragma unroll
  for (int i = 0; i < 8; ++i) r.u[i] = 0u;
  return r.v;
}

// ---------------------------------------------------------------------------
// Generic tiled GEMM: C[z] (MxN) = A[z] (MxK, row-major bf16) * B[z] (KxN row-major)
// BSRC: 0 = B is bf16 (Kact guard, zero-fill), 1 = B is f32 with fused LayerNorm
// MODE: 0 = store bf16, 1 = store f32, 2 = store f32 + residual(f32)
// z decomposed as zo = z / zh, zi = z % zh for (batch, head) strides.
// Tile: 64(M) x 128(N) x 32(K); 8 waves = 4(M-strips) x 2(N-halves); wave: 16x64.
// Double-buffered LDS staging: one barrier/K-step; next tile's global loads
// issue before the current tile's WMMA chain (latency hidden by compute).
// ---------------------------------------------------------------------------
template<int BSRC, int MODE>
__global__ __launch_bounds__(256) void gemm_nn_kernel(
    const bf16* __restrict__ A, long sAo, long sAi, int lda, int M,
    const void* __restrict__ Bsrc, long sBo, long sBi, int ldb, int Kact, int KT,
    const float* __restrict__ mu, const float* __restrict__ rstd, long sStat,
    const float* __restrict__ lnw, const float* __restrict__ lnb,
    void* __restrict__ Cdst, long sCo, long sCi, int ldc,
    const float* __restrict__ Rsd, long sRo,
    int zh, int N)
{
  __shared__ __align__(16) unsigned int ldsBu[2 * 2560];  // 2 x (128 n-rows, stride 40 bf16)

  const int z  = blockIdx.z;
  const int zo = z / zh;
  const int zi = z - zo * zh;
  const int n0 = blockIdx.x * 128;
  const int m0 = blockIdx.y * 64;
  const int tid  = threadIdx.x;
  const int lane = tid & 31;
  const int wave = tid >> 5;
  const int wm = wave & 3;
  const int wn = wave >> 2;
  const int l15 = lane & 15;
  const bool lo = lane < 16;
  const int aKoff = lo ? 0 : 8;
  const int bKoff = lo ? 0 : 8;
  const int mRow  = m0 + wm * 16 + l15;
  const bf16* Az = A + zo * sAo + zi * sAi;

  const int kk = tid & 15;   // k-pair index in staging
  const int nc = tid >> 4;   // n-chunk (8 wide) in staging
  const int nb = nc * 8;

  // hoisted per-pixel LN stats (depend only on n)
  float muv[8], rsv[8];
  const float* BzF = (const float*)Bsrc + zo * sBo + zi * sBi;
  const bf16*  BzH = (const bf16*)Bsrc  + zo * sBo + zi * sBi;
  if (BSRC == 1) {
    const float* muz = mu   + zo * sStat + n0 + nb;
    const float* rsz = rstd + zo * sStat + n0 + nb;
    #pragma unroll
    for (int i = 0; i < 8; ++i) { muv[i] = muz[i]; rsv[i] = rsz[i]; }
  }

  // staging registers (double-buffer pipeline)
  float sf0[8], sf1[8], slw0 = 0.f, slw1 = 0.f, slb0 = 0.f, slb1 = 0.f;
  union UB { v4u u; bf16 h[8]; } sb0, sb1;

  auto fetch = [&](int kt) {
    const int k0 = kt * 32 + 2 * kk;
    if (BSRC == 1) {
      const float* p0 = BzF + (long)k0 * ldb + n0 + nb;
      const float* p1 = p0 + ldb;
      #pragma unroll
      for (int i = 0; i < 8; ++i) { sf0[i] = p0[i]; sf1[i] = p1[i]; }
      slw0 = lnw[k0];     slb0 = lnb[k0];
      slw1 = lnw[k0 + 1]; slb1 = lnb[k0 + 1];
    } else {
      if (k0 < Kact) {
        sb0.u = *(const v4u*)(BzH + (long)k0 * ldb + n0 + nb);
        sb1.u = *(const v4u*)(BzH + (long)(k0 + 1) * ldb + n0 + nb);
      } else {
        #pragma unroll
        for (int i = 0; i < 8; ++i) { sb0.h[i] = (bf16)0.f; sb1.h[i] = (bf16)0.f; }
      }
    }
  };

  fetch(0);

  v8f acc[4];
  #pragma unroll
  for (int t = 0; t < 4; ++t)
    #pragma unroll
    for (int i = 0; i < 8; ++i) acc[t][i] = 0.f;

  for (int kt = 0; kt < KT; ++kt) {
    const int buf = kt & 1;
    unsigned int* dst = ldsBu + buf * 2560;
    // ---- commit staged tile to LDS (transposed [n][k], k-pairs as u32) ----
    if (BSRC == 1) {
      #pragma unroll
      for (int i = 0; i < 8; ++i) {
        union { unsigned int u; bf16 h[2]; } pk;
        pk.h[0] = (bf16)((sf0[i] - muv[i]) * rsv[i] * slw0 + slb0);
        pk.h[1] = (bf16)((sf1[i] - muv[i]) * rsv[i] * slw1 + slb1);
        dst[(nb + i) * 20 + kk] = pk.u;
      }
    } else {
      #pragma unroll
      for (int i = 0; i < 8; ++i) {
        union { unsigned int u; bf16 h[2]; } pk;
        pk.h[0] = sb0.h[i]; pk.h[1] = sb1.h[i];
        dst[(nb + i) * 20 + kk] = pk.u;
      }
    }
    __syncthreads();

    // ---- issue next tile's global loads before the WMMA chain ----
    if (kt + 1 < KT) fetch(kt + 1);
    if (kt + 2 < KT) {
      const int k0 = (kt + 2) * 32 + 2 * kk;
      if (BSRC == 1) __builtin_prefetch(BzF + (long)k0 * ldb + n0 + nb, 0, 1);
      else if (k0 < Kact) __builtin_prefetch(BzH + (long)k0 * ldb + n0 + nb, 0, 1);
    }

    // ---- A fragment straight from global (weights are L2-resident) ----
    const int kb = kt * 32;
    v16bf af;
    if (mRow < M) {
      const bf16* ap = Az + (long)mRow * lda + kb + aKoff;
      af = ld_frag_g(ap, ap + 16);
    } else {
      af = zfrag();
    }
    // ---- 4 WMMAs over the 16x64 wave tile ----
    const bf16* ldsB = (const bf16*)(ldsBu + buf * 2560);
    #pragma unroll
    for (int nt = 0; nt < 4; ++nt) {
      const bf16* bp = ldsB + (wn * 64 + nt * 16 + l15) * 40 + bKoff;
      union { v4u u[2]; v16bf v; } bfv;
      bfv.u[0] = *(const v4u*)bp;
      bfv.u[1] = *(const v4u*)(bp + 16);
      acc[nt] = WMMA_BF16(af, bfv.v, acc[nt]);
    }
  }

  // ---- epilogue ----
  const int rBase = lo ? 0 : 8;
  #pragma unroll
  for (int nt = 0; nt < 4; ++nt) {
    const int col = n0 + wn * 64 + nt * 16 + l15;
    #pragma unroll
    for (int r = 0; r < 8; ++r) {
      const int row = m0 + wm * 16 + rBase + r;
      if (row < M) {
        const long off = zo * sCo + zi * sCi + (long)row * ldc + col;
        float v = acc[nt][r];
        if (MODE == 2) v += Rsd[zo * sRo + (long)row * ldc + col];
        if (MODE == 0) ((bf16*)Cdst)[off] = (bf16)v;
        else           ((float*)Cdst)[off] = v;
      }
    }
  }
}

// ---------------------------------------------------------------------------
// attn partial: (q . k^T) * rq * rk * temperature   (per b,head; K-split)
// M=N=48, K slice of 16384; both operands K-contiguous -> direct fragments.
// grid = (32 z, 8 K-slices); 9 waves, one 16x16 D tile each.
// Deterministic split-K: each slice writes its own scaled partial tile.
// ---------------------------------------------------------------------------
__global__ __launch_bounds__(288) void attn_qk_kernel(
    const bf16* __restrict__ qkv, const float* __restrict__ rcpn,
    const float* __restrict__ temp, float* __restrict__ araw, int N, int KS)
{
  const int z  = blockIdx.x;        // b*4 + h
  const int ks = blockIdx.y;        // K-slice
  const int b = z >> 2, h = z & 3;
  const bf16* qb  = qkv + ((long)b * 576 + (long)h * 48) * N;
  const bf16* kbp = qb + (long)192 * N;
  const int wave = threadIdx.x / 32;
  const int lane = threadIdx.x & 31;
  const int ti = wave / 3, tj = wave - ti * 3;
  const int l15 = lane & 15;
  const bool lo = lane < 16;
  const int aKoff = lo ? 0 : 8;
  const int bKoff = lo ? 0 : 8;
  const bf16* qp = qb  + (long)(ti * 16 + l15) * N;
  const bf16* kp = kbp + (long)(tj * 16 + l15) * N;

  v8f acc;
  #pragma unroll
  for (int i = 0; i < 8; ++i) acc[i] = 0.f;

  const int kbeg = ks * KS, kend = kbeg + KS;
  #pragma unroll 8
  for (int kb = kbeg; kb < kend; kb += 32) {
    v16bf af = ld_frag_g(qp + kb + aKoff, qp + kb + aKoff + 16);
    v16bf bf = ld_frag_g(kp + kb + bKoff, kp + kb + bKoff + 16);
    acc = WMMA_BF16(af, bf, acc);
  }

  const int rBase = lo ? 0 : 8;
  const float tm = temp[h];
  #pragma unroll
  for (int r = 0; r < 8; ++r) {
    const int row = ti * 16 + rBase + r;
    const int col = tj * 16 + l15;
    float v = acc[r] * rcpn[b * 384 + h * 48 + row]
                     * rcpn[b * 384 + 192 + h * 48 + col] * tm;
    araw[(((long)ks * 32 + z) * 48 + row) * 48 + col] = v;
  }
}

// ---------------------------------------------------------------------------
// per-pixel LayerNorm stats over C channels of (B,C,HW) f32
// ---------------------------------------------------------------------------
__global__ __launch_bounds__(256) void ln_stats_kernel(
    const float* __restrict__ x, float* __restrict__ mu, float* __restrict__ rstd,
    int C, int HW, long total)
{
  long i = (long)blockIdx.x * 256 + threadIdx.x;
  if (i >= total) return;
  long b = i / HW;
  int  n = (int)(i - b * HW);
  const float* p = x + b * (long)C * HW + n;
  float s = 0.f, s2 = 0.f;
  for (int c = 0; c < C; ++c) {
    float v = p[(long)c * HW];
    s += v; s2 += v * v;
  }
  float m = s / C;
  float var = s2 / C - m * m;
  mu[i] = m;
  rstd[i] = rsqrtf(var + 1e-5f);
}

// ---------------------------------------------------------------------------
// depthwise 3x3, SAME, bf16 in/out, f32 accumulate
// ---------------------------------------------------------------------------
__global__ __launch_bounds__(256) void dwconv3x3_kernel(
    const bf16* __restrict__ in, const float* __restrict__ w,
    bf16* __restrict__ out, int Ch, int H, int W, long total)
{
  long idx = (long)blockIdx.x * 256 + threadIdx.x;
  if (idx >= total) return;
  int xw = (int)(idx % W);
  long t = idx / W;
  int yh = (int)(t % H); t /= H;
  int c  = (int)(t % Ch);
  const float* wc = w + c * 9;
  const bf16* p = in + (idx - xw - (long)yh * W);
  float s = 0.f;
  #pragma unroll
  for (int dy = -1; dy <= 1; ++dy) {
    int yy = yh + dy;
    if (yy < 0 || yy >= H) continue;
    #pragma unroll
    for (int dx = -1; dx <= 1; ++dx) {
      int xx = xw + dx;
      if (xx < 0 || xx >= W) continue;
      s += wc[(dy + 1) * 3 + (dx + 1)] * (float)p[(long)yy * W + xx];
    }
  }
  out[idx] = (bf16)s;
}

// ---------------------------------------------------------------------------
// fused depthwise 3x3 on pin (1020 ch) + GELU gate -> g (512 ch, ch>=510 zero)
// ---------------------------------------------------------------------------
__global__ __launch_bounds__(256) void dw_gelu_kernel(
    const bf16* __restrict__ in, const float* __restrict__ w,
    bf16* __restrict__ g, int H, int W, long total)
{
  long idx = (long)blockIdx.x * 256 + threadIdx.x;
  if (idx >= total) return;
  int xw = (int)(idx % W);
  long t = idx / W;
  int yh = (int)(t % H); t /= H;
  int c  = (int)(t % 512);
  int b  = (int)(t / 512);
  if (c >= 510) { g[idx] = (bf16)0.f; return; }
  const long HW = (long)H * W;
  const bf16* p1 = in + ((long)b * 1020 + c) * HW;
  const bf16* p2 = p1 + (long)510 * HW;
  const float* w1 = w + c * 9;
  const float* w2 = w + (c + 510) * 9;
  float s1 = 0.f, s2 = 0.f;
  #pragma unroll
  for (int dy = -1; dy <= 1; ++dy) {
    int yy = yh + dy;
    if (yy < 0 || yy >= H) continue;
    #pragma unroll
    for (int dx = -1; dx <= 1; ++dx) {
      int xx = xw + dx;
      if (xx < 0 || xx >= W) continue;
      long o = (long)yy * W + xx;
      float wv1 = w1[(dy + 1) * 3 + (dx + 1)];
      float wv2 = w2[(dy + 1) * 3 + (dx + 1)];
      s1 += wv1 * (float)p1[o];
      s2 += wv2 * (float)p2[o];
    }
  }
  float ge = 0.5f * s1 * (1.f + erff(s1 * 0.70710678118654752f));
  g[idx] = (bf16)(ge * s2);
}

// ---------------------------------------------------------------------------
// 1/max(||row||,eps) for q,k rows (row length N), one workgroup per row
// ---------------------------------------------------------------------------
__global__ __launch_bounds__(256) void row_rcpnorm_kernel(
    const bf16* __restrict__ qkvdw, float* __restrict__ rcpn, int N)
{
  const int r = blockIdx.x;          // 0..3071 : b*384 + chan(0..383)
  const int b = r / 384;
  const int ch = r - b * 384;
  const bf16* p = qkvdw + ((long)b * 576 + ch) * N;
  float s = 0.f;
  for (int i = threadIdx.x; i < N; i += 256) {
    float v = (float)p[i];
    s += v * v;
  }
  __shared__ float red[256];
  red[threadIdx.x] = s;
  __syncthreads();
  for (int off = 128; off > 0; off >>= 1) {
    if (threadIdx.x < off) red[threadIdx.x] += red[threadIdx.x + off];
    __syncthreads();
  }
  if (threadIdx.x == 0) rcpn[r] = 1.f / fmaxf(sqrtf(red[0]), 1e-12f);
}

// ---------------------------------------------------------------------------
// sum K-split partials, softmax over 48, output bf16 padded to 64 (zeros)
// ---------------------------------------------------------------------------
__global__ __launch_bounds__(256) void softmax48_kernel(
    const float* __restrict__ a, bf16* __restrict__ o, int rows,
    int nPart, long pStride)
{
  int r = blockIdx.x * 256 + threadIdx.x;
  if (r >= rows) return;
  float v48[48];
  for (int d = 0; d < 48; ++d) {
    float s = 0.f;
    for (int p = 0; p < nPart; ++p) s += a[p * pStride + (long)r * 48 + d];
    v48[d] = s;
  }
  float mx = -1e30f;
  for (int d = 0; d < 48; ++d) mx = fmaxf(mx, v48[d]);
  float s = 0.f;
  for (int d = 0; d < 48; ++d) { v48[d] = __expf(v48[d] - mx); s += v48[d]; }
  float inv = 1.f / s;
  bf16* q = o + (long)r * 64;
  for (int d = 0; d < 48; ++d) q[d] = (bf16)(v48[d] * inv);
  for (int d = 48; d < 64; ++d) q[d] = (bf16)0.f;
}

// ---------------------------------------------------------------------------
// f32 -> bf16 weight conversion with column padding
// ---------------------------------------------------------------------------
__global__ __launch_bounds__(256) void cvt_w_kernel(
    const float* __restrict__ s, bf16* __restrict__ d,
    int cols, int ldd, long total)
{
  long i = (long)blockIdx.x * 256 + threadIdx.x;
  if (i >= total) return;
  long r = i / ldd;
  int  c = (int)(i - r * ldd);
  d[i] = (c < cols) ? (bf16)s[r * (long)cols + c] : (bf16)0.f;
}

// ===========================================================================
extern "C" void kernel_launch(void* const* d_in, const int* in_sizes, int n_in,
                              void* d_out, int out_size, void* d_ws, size_t ws_size,
                              hipStream_t stream) {
  (void)in_sizes; (void)n_in; (void)out_size; (void)ws_size;
  const float* x      = (const float*)d_in[0];
  const float* ln1w   = (const float*)d_in[1];
  const float* ln1b   = (const float*)d_in[2];
  const float* qkvw   = (const float*)d_in[3];
  const float* qkvdwW = (const float*)d_in[4];
  const float* temp   = (const float*)d_in[5];
  const float* projw  = (const float*)d_in[6];
  const float* ln2w   = (const float*)d_in[7];
  const float* ln2b   = (const float*)d_in[8];
  const float* pinw   = (const float*)d_in[9];
  const float* pindwW = (const float*)d_in[10];
  const float* poutw  = (const float*)d_in[11];
  float* out = (float*)d_out;

  const int  N = 16384, C = 192, B = 8, H = 128, W = 128;
  const long CN = (long)C * N;

  char* ws = (char*)d_ws;
  size_t o = 0;
  auto alloc = [&](size_t bytes) -> size_t {
    size_t r = o; o += (bytes + 255) & ~(size_t)255; return r;
  };
  const size_t o_wq  = alloc((size_t)576 * 192 * 2);
  const size_t o_wp  = alloc((size_t)192 * 192 * 2);
  const size_t o_wi  = alloc((size_t)1020 * 192 * 2);
  const size_t o_wo  = alloc((size_t)192 * 512 * 2);
  const size_t o_mu1 = alloc((size_t)B * N * 4);
  const size_t o_rs1 = alloc((size_t)B * N * 4);
  const size_t o_mu2 = alloc((size_t)B * N * 4);
  const size_t o_rs2 = alloc((size_t)B * N * 4);
  const size_t o_rcp = alloc((size_t)3072 * 4);
  const size_t o_araw= alloc((size_t)8 * 32 * 48 * 48 * 4);   // 8 K-split partials
  const size_t o_asm = alloc((size_t)32 * 48 * 64 * 2);
  const size_t o_y   = alloc((size_t)B * CN * 4);          // y f32, live to end
  // shared big region with lifetime aliasing
  const size_t big   = o;
  const size_t o_qkv = big;                   // qkv pre-dw bf16 (151MB)
  const size_t o_dw  = big + 150994944UL;     // qkv post-dw bf16 (151MB)
  const size_t o_att = big;                   // attn out bf16 (50MB), after o_qkv dead
  const size_t o_g   = big;                   // gated bf16 (134MB), after o_att dead
  const size_t o_pin = big + 134217728UL;     // pin bf16 (267MB), after o_dw dead

  bf16*  wq   = (bf16*)(ws + o_wq);
  bf16*  wp   = (bf16*)(ws + o_wp);
  bf16*  wi   = (bf16*)(ws + o_wi);
  bf16*  wo   = (bf16*)(ws + o_wo);
  float* mu1  = (float*)(ws + o_mu1);
  float* rs1  = (float*)(ws + o_rs1);
  float* mu2  = (float*)(ws + o_mu2);
  float* rs2  = (float*)(ws + o_rs2);
  float* rcp  = (float*)(ws + o_rcp);
  float* araw = (float*)(ws + o_araw);
  bf16*  asmv = (bf16*)(ws + o_asm);
  float* yb   = (float*)(ws + o_y);
  bf16*  qkvp = (bf16*)(ws + o_qkv);
  bf16*  qdw  = (bf16*)(ws + o_dw);
  bf16*  attb = (bf16*)(ws + o_att);
  bf16*  gb   = (bf16*)(ws + o_g);
  bf16*  pinb = (bf16*)(ws + o_pin);

  // 1) weights -> bf16 (pout padded 510->512)
  cvt_w_kernel<<<432, 256, 0, stream>>>(qkvw,  wq, 192, 192, (long)576 * 192);
  cvt_w_kernel<<<144, 256, 0, stream>>>(projw, wp, 192, 192, (long)192 * 192);
  cvt_w_kernel<<<765, 256, 0, stream>>>(pinw,  wi, 192, 192, (long)1020 * 192);
  cvt_w_kernel<<<384, 256, 0, stream>>>(poutw, wo, 510, 512, (long)192 * 512);

  // 2) LN1 stats on x
  ln_stats_kernel<<<512, 256, 0, stream>>>(x, mu1, rs1, C, N, (long)B * N);

  // 3) qkv = qkv_w * LN1(x)   (fused-LN B operand)  -> qkvp bf16
  gemm_nn_kernel<1, 0><<<dim3(128, 9, 8), 256, 0, stream>>>(
      wq, 0, 0, 192, 576,
      x, CN, 0, N, 192, 6,
      mu1, rs1, (long)N, ln1w, ln1b,
      qkvp, (long)576 * N, 0, N,
      nullptr, 0, 1, N);

  // 4) depthwise 3x3 on qkv
  dwconv3x3_kernel<<<294912, 256, 0, stream>>>(qkvp, qkvdwW, qdw, 576, H, W,
                                               (long)B * 576 * N);

  // 5) row L2 norms of q,k
  row_rcpnorm_kernel<<<3072, 256, 0, stream>>>(qdw, rcp, N);

  // 6) attn partials = (q.k^T) * rq * rk * temperature, K split 8 ways
  attn_qk_kernel<<<dim3(32, 8), 288, 0, stream>>>(qdw, rcp, temp, araw, N, 2048);

  // 7) sum partials + softmax(48) -> bf16, K-padded to 64
  softmax48_kernel<<<6, 256, 0, stream>>>(araw, asmv, 1536, 8, (long)32 * 48 * 48);

  // 8) out = attn * v   (per b,head; M=48, Kpad=64, Kact=48) -> attb bf16
  gemm_nn_kernel<0, 0><<<dim3(128, 1, 32), 256, 0, stream>>>(
      asmv, 12288, 3072, 64, 48,
      qdw + (long)384 * N, (long)576 * N, (long)48 * N, N, 48, 2,
      nullptr, nullptr, 0, nullptr, nullptr,
      attb, (long)192 * N, (long)48 * N, N,
      nullptr, 0, 4, N);

  // 9) y = x + proj_w * out   -> f32
  gemm_nn_kernel<0, 2><<<dim3(128, 3, 8), 256, 0, stream>>>(
      wp, 0, 0, 192, 192,
      attb, (long)192 * N, 0, N, 192, 6,
      nullptr, nullptr, 0, nullptr, nullptr,
      yb, CN, 0, N,
      x, CN, 1, N);

  // 10) LN2 stats on y
  ln_stats_kernel<<<512, 256, 0, stream>>>(yb, mu2, rs2, C, N, (long)B * N);

  // 11) pin = pin_w * LN2(y)  -> bf16 (1020 ch)
  gemm_nn_kernel<1, 0><<<dim3(128, 16, 8), 256, 0, stream>>>(
      wi, 0, 0, 192, 1020,
      yb, CN, 0, N, 192, 6,
      mu2, rs2, (long)N, ln2w, ln2b,
      pinb, (long)1020 * N, 0, N,
      nullptr, 0, 1, N);

  // 12) depthwise 3x3 + GELU gate -> g bf16 (512 ch, padded)
  dw_gelu_kernel<<<262144, 256, 0, stream>>>(pinb, pindwW, gb, H, W,
                                             (long)B * 512 * N);

  // 13) out = y + pout_w * g
  gemm_nn_kernel<0, 2><<<dim3(128, 3, 8), 256, 0, stream>>>(
      wo, 0, 0, 512, 192,
      gb, (long)512 * N, 0, N, 512, 16,
      nullptr, nullptr, 0, nullptr, nullptr,
      out, CN, 0, N,
      yb, CN, 1, N);
}
